// RationaleSelectorModel_20564303413304
// MI455X (gfx1250) — matrix operand order repaired and checked by reading, not compiled, not run on hardware
//
#include <hip/hip_runtime.h>
#include <hip/hip_bf16.h>
#include <math.h>

#define BB 16
#define TT 2048
#define DD 768
#define HH 1024
#define SS 5
#define LDA 776   // padded LDS row stride (bf16 elems): 1552B, 16B-aligned, breaks bank alignment

typedef __attribute__((ext_vector_type(16))) __bf16 v16bf;
typedef __attribute__((ext_vector_type(8)))  float  v8f;

union FragBF { v16bf v; unsigned int u[8]; };

// ---------------------------------------------------------------------------
// w1 f32 -> bf16
// ---------------------------------------------------------------------------
__global__ void k_w1cvt(const float* __restrict__ w1, __bf16* __restrict__ w1bf, int n) {
  int i = blockIdx.x * blockDim.x + threadIdx.x;
  if (i < n) w1bf[i] = (__bf16)w1[i];
}

// ---------------------------------------------------------------------------
// Fused LN + GEMM + GELU·w2 contraction:
//   scores[tok] = sum_h gelu( LN(emb*attn)[tok] @ w1^T + b1 )[h] * w2[h]
// Block = 16 tokens x full H, 8 waves.
// Phase 1: each 16-lane half-wave LayerNorms one token row -> bf16 in LDS.
// Phase 2: K-loop, A fragments from LDS (ds_load), B from L2-resident w1bf,
//          8 v_wmma_f32_16x16x32_bf16 per wave per K-step.
// Phase 3: gelu epilogue, 16-lane shfl reduction, cross-wave LDS reduction,
//          direct store of the 16 scores (no atomics, no pre-zero).
// ---------------------------------------------------------------------------
__global__ __launch_bounds__(256) void k_gemm(const float* __restrict__ emb,
                                              const float* __restrict__ attn,
                                              const float* __restrict__ ln_g,
                                              const float* __restrict__ ln_b,
                                              const __bf16* __restrict__ w1bf,
                                              const float* __restrict__ b1,
                                              const float* __restrict__ w2,
                                              float* __restrict__ scores_raw) {
  const int tokBase = blockIdx.x * 16;
  const int wave = threadIdx.x >> 5;
  const int lane = threadIdx.x & 31;
  const int half = lane >> 4;
  const int nm   = lane & 15;
  const int hBase = wave * 128;

  __shared__ __bf16 s_a[16 * LDA];
  __shared__ float  s_sc[8][16];

  // ---- Phase 1: LayerNorm 16 token rows into LDS (bf16) ------------------
  {
    const int row = 2 * wave + half;          // 0..15, one row per 16-lane group
    const int tok = tokBase + row;
    const float a = attn[tok];
    const float* erow = emb + (size_t)tok * DD;
    float ev[48];
    float sum = 0.0f, sq = 0.0f;
#pragma unroll
    for (int i = 0; i < 48; ++i) {
      float x = erow[nm + 16 * i] * a;
      ev[i] = x; sum += x; sq += x * x;
    }
    sum += __shfl_xor(sum, 1, 16); sq += __shfl_xor(sq, 1, 16);
    sum += __shfl_xor(sum, 2, 16); sq += __shfl_xor(sq, 2, 16);
    sum += __shfl_xor(sum, 4, 16); sq += __shfl_xor(sq, 4, 16);
    sum += __shfl_xor(sum, 8, 16); sq += __shfl_xor(sq, 8, 16);
    const float mu   = sum * (1.0f / DD);
    const float var  = sq * (1.0f / DD) - mu * mu;
    const float rstd = rsqrtf(var + 1e-5f);
    __bf16* srow = s_a + row * LDA;
#pragma unroll
    for (int i = 0; i < 48; ++i) {
      int d = nm + 16 * i;
      srow[d] = (__bf16)((ev[i] - mu) * rstd * ln_g[d] + ln_b[d]);
    }
  }
  __syncthreads();

  // ---- Phase 2: WMMA K-loop ----------------------------------------------
  // K-offset pattern (elements, even -> dword/b128-mergeable) for the bf16
  // A 16x32 fragment (ISA 7.12.2 16-bit A layout) and B 32x16 fragment.
  int ka[8], kb[8];
#pragma unroll
  for (int v = 0; v < 8; ++v) {
    ka[v] = ((v >> 2) << 4) + 8 * half + 2 * (v & 3);
    kb[v] = ((v >> 2) << 4) + 8 * half + 2 * (v & 3);
  }

  v8f c[8];
#pragma unroll
  for (int t = 0; t < 8; ++t)
#pragma unroll
    for (int e = 0; e < 8; ++e) c[t][e] = 0.0f;

  const __bf16* arow  = s_a  + (size_t)nm * LDA;            // LDS, row = token
  const __bf16* brow0 = w1bf + (size_t)(hBase + nm) * DD;   // L2,  row = h channel

  for (int k0 = 0; k0 < DD; k0 += 32) {
    FragBF fa;
#pragma unroll
    for (int v = 0; v < 8; ++v)
      fa.u[v] = *(const unsigned int*)(arow + k0 + ka[v]);
#pragma unroll
    for (int t = 0; t < 8; ++t) {
      const __bf16* brow = brow0 + (size_t)t * 16 * DD;
      FragBF fb;
#pragma unroll
      for (int v = 0; v < 8; ++v)
        fb.u[v] = *(const unsigned int*)(brow + k0 + kb[v]);
      c[t] = __builtin_amdgcn_wmma_f32_16x16x32_bf16(false, fa.v, false, fb.v,
                                                     (short)0, c[t], false, false);
    }
  }

  // ---- Phase 3: epilogue --------------------------------------------------
  // C layout: lane l, vgpr r -> M = r + 8*(l>>4), N = l&15.
  float part[8];
#pragma unroll
  for (int r = 0; r < 8; ++r) part[r] = 0.0f;
#pragma unroll
  for (int t = 0; t < 8; ++t) {
    const int hcol = hBase + t * 16 + nm;
    const float bb = b1[hcol];
    const float ww = w2[hcol];
#pragma unroll
    for (int r = 0; r < 8; ++r) {
      float hv = c[t][r] + bb;
      float ge = 0.5f * hv * (1.0f + erff(hv * 0.70710678118f));
      part[r] += ge * ww;
    }
  }
#pragma unroll
  for (int r = 0; r < 8; ++r) {
    float v = part[r];
    v += __shfl_xor(v, 1, 16);
    v += __shfl_xor(v, 2, 16);
    v += __shfl_xor(v, 4, 16);
    v += __shfl_xor(v, 8, 16);
    if (nm == 0) s_sc[wave][8 * half + r] = v;
  }
  __syncthreads();
  if (threadIdx.x < 16) {
    float sum = 0.0f;
#pragma unroll
    for (int w = 0; w < 8; ++w) sum += s_sc[w][threadIdx.x];
    scores_raw[tokBase + threadIdx.x] = sum;
  }
}

// ---------------------------------------------------------------------------
// entmax 1.5 along T for one batch row per block. Bitonic sort + cumsums.
// ---------------------------------------------------------------------------
__global__ __launch_bounds__(256) void k_entmax(const float* __restrict__ scores_raw,
                                                const float* __restrict__ attn,
                                                const float* __restrict__ b2,
                                                float* __restrict__ scores_mask,
                                                float* __restrict__ teff,
                                                float* __restrict__ z_out) {
  const int b = blockIdx.x;
  const int tid = threadIdx.x;
  __shared__ float s_z[TT];
  __shared__ float s_c1[TT];
  __shared__ float s_c2[TT];
  __shared__ float s_t1[256];
  __shared__ float s_t2[256];
  __shared__ float s_red[256];
  __shared__ int   s_support;
  __shared__ float s_tau;

  const float bias2 = b2[0];
  float asum = 0.0f;
  for (int t = tid; t < TT; t += 256) {
    float a  = attn[b * TT + t];
    float sc = scores_raw[b * TT + t] + bias2;
    sc = (a == 0.0f) ? -1e9f : sc;
    scores_mask[b * TT + t] = sc;
    s_z[t] = 0.5f * sc;            // entmax15 works on z/2 (TAU == 1)
    asum += a;
  }
  s_red[tid] = asum;
  __syncthreads();
  for (int off = 128; off > 0; off >>= 1) {
    if (tid < off) s_red[tid] += s_red[tid + off];
    __syncthreads();
  }
  if (tid == 0) teff[b] = s_red[0];

  // bitonic sort, descending
  for (int kk = 2; kk <= TT; kk <<= 1) {
    for (int jj = kk >> 1; jj > 0; jj >>= 1) {
      __syncthreads();
      for (int i = tid; i < TT; i += 256) {
        int ixj = i ^ jj;
        if (ixj > i) {
          float av = s_z[i], bv = s_z[ixj];
          bool sw = ((i & kk) == 0) ? (av < bv) : (av > bv);
          if (sw) { s_z[i] = bv; s_z[ixj] = av; }
        }
      }
    }
  }
  __syncthreads();

  // cumsum of z and z^2: serial over each thread's 8-chunk + scan of totals
  float r1 = 0.0f, r2 = 0.0f;
  const int base = tid * 8;
  for (int e = 0; e < 8; ++e) {
    float zz = s_z[base + e];
    r1 += zz; r2 += zz * zz;
    s_c1[base + e] = r1; s_c2[base + e] = r2;
  }
  s_t1[tid] = r1; s_t2[tid] = r2;
  __syncthreads();
  for (int off = 1; off < 256; off <<= 1) {
    float a1 = (tid >= off) ? s_t1[tid - off] : 0.0f;
    float a2 = (tid >= off) ? s_t2[tid - off] : 0.0f;
    __syncthreads();
    s_t1[tid] += a1; s_t2[tid] += a2;
    __syncthreads();
  }
  const float off1 = s_t1[tid] - r1;
  const float off2 = s_t2[tid] - r2;
  for (int e = 0; e < 8; ++e) { s_c1[base + e] += off1; s_c2[base + e] += off2; }
  if (tid == 0) s_support = 0;
  __syncthreads();

  int cnt = 0;
  for (int j = tid; j < TT; j += 256) {
    float rho  = (float)(j + 1);
    float mean = s_c1[j] / rho;
    float msq  = s_c2[j] / rho;
    float ss   = rho * (msq - mean * mean);
    float delta = (1.0f - ss) / rho;
    float tau   = mean - sqrtf(fmaxf(delta, 0.0f));
    if (tau <= s_z[j]) cnt++;
  }
  atomicAdd(&s_support, cnt);
  __syncthreads();
  if (tid == 0) {
    int j = s_support - 1; if (j < 0) j = 0;
    float rho  = (float)(j + 1);
    float mean = s_c1[j] / rho;
    float msq  = s_c2[j] / rho;
    float ss   = rho * (msq - mean * mean);
    float delta = (1.0f - ss) / rho;
    s_tau = mean - sqrtf(fmaxf(delta, 0.0f));
  }
  __syncthreads();
  const float tau_star = s_tau;
  for (int t = tid; t < TT; t += 256) {
    float a  = attn[b * TT + t];
    float zh = 0.5f * scores_mask[b * TT + t];
    float p  = fmaxf(zh - tau_star, 0.0f);
    z_out[b * TT + t] = p * p * a;
  }
}

// ---------------------------------------------------------------------------
// Probabilistic top-k: one block per (s,b). g == straight-through h.
// ---------------------------------------------------------------------------
__global__ __launch_bounds__(256) void k_topk(const float* __restrict__ scores_mask,
                                              const float* __restrict__ attn,
                                              const float* __restrict__ u,
                                              const float* __restrict__ teff,
                                              float* __restrict__ g_out,
                                              float* __restrict__ hsum) {
  const int sb = blockIdx.x;           // s*BB + b
  const int s = sb / BB, b = sb % BB;
  const int tid = threadIdx.x;
  __shared__ float s_v[TT];
  __shared__ int   s_i[TT];
  __shared__ float s_red[256];

  for (int t = tid; t < TT; t += 256) {
    float a   = attn[b * TT + t];
    float sm  = scores_mask[b * TT + t] * a;   // scores * attn
    float uu  = u[(size_t)sb * TT + t];
    float gum = -logf(-logf(uu + 1e-6f) + 1e-6f);
    s_v[t] = sm + gum;
    s_i[t] = t;
  }
  // bitonic: descending by value, index-ascending tiebreak (stable argsort)
  for (int kk = 2; kk <= TT; kk <<= 1) {
    for (int jj = kk >> 1; jj > 0; jj >>= 1) {
      __syncthreads();
      for (int i = tid; i < TT; i += 256) {
        int ixj = i ^ jj;
        if (ixj > i) {
          float av = s_v[i], bv = s_v[ixj];
          int   ai = s_i[i], bi = s_i[ixj];
          bool iAfter = (av < bv) || (av == bv && ai > bi);
          bool sw = ((i & kk) == 0) ? iAfter : !iAfter;
          if (sw) { s_v[i] = bv; s_v[ixj] = av; s_i[i] = bi; s_i[ixj] = ai; }
        }
      }
    }
  }
  __syncthreads();
  float rho = 0.1f + 0.2f * (float)s;          // linspace(0.1, 0.9, 5)
  int k = (int)floorf(rho * teff[b]);
  if (k < 1) k = 1;

  float* grow = g_out + (size_t)sb * TT;
  for (int t = tid; t < TT; t += 256) grow[t] = 0.0f;
  __syncthreads();
  float hs = 0.0f;
  for (int j = tid; j < k; j += 256) {
    int t0 = s_i[j];
    float hv = attn[b * TT + t0];
    grow[t0] = hv;
    hs += hv;
  }
  s_red[tid] = hs;
  __syncthreads();
  for (int off = 128; off > 0; off >>= 1) {
    if (tid < off) s_red[tid] += s_red[tid + off];
    __syncthreads();
  }
  if (tid == 0) hsum[sb] = s_red[0];
}

// ---------------------------------------------------------------------------
// Fused pooling: one block per batch row computes full_rep AND all 5
// pred_reps in a single pass over the gathered emb_table rows (6x traffic
// reduction). Weights staged in LDS; next-row prefetch via global_prefetch.
// ---------------------------------------------------------------------------
__global__ __launch_bounds__(256) void k_pool(const int* __restrict__ ids,
                                              const float* __restrict__ attn,
                                              const float* __restrict__ g_out,
                                              const float* __restrict__ emb_table,
                                              const float* __restrict__ teff,
                                              const float* __restrict__ hsum,
                                              float* __restrict__ full,
                                              float* __restrict__ pred) {
  const int b = blockIdx.x, tid = threadIdx.x;
  __shared__ float s_w0[TT];        // attn weights
  __shared__ float s_g[SS][TT];     // top-k indicator weights per sweep
  __shared__ int   s_id[TT];

  for (int t = tid; t < TT; t += 256) {
    s_w0[t] = attn[b * TT + t];
    s_id[t] = ids[b * TT + t];
#pragma unroll
    for (int s = 0; s < SS; ++s)
      s_g[s][t] = g_out[((size_t)s * BB + b) * TT + t];
  }
  __syncthreads();

  float f0 = 0, f1 = 0, f2 = 0;
  float p0[SS], p1[SS], p2[SS];
#pragma unroll
  for (int s = 0; s < SS; ++s) { p0[s] = 0; p1[s] = 0; p2[s] = 0; }

  for (int t = 0; t < TT; ++t) {
    const float wa = s_w0[t];
    if (wa == 0.0f) continue;                  // contiguous mask -> uniform branch
    const float* row = emb_table + (size_t)s_id[t] * DD;
    if (t + 1 < TT && s_w0[t + 1] != 0.0f)     // hide gather latency
      __builtin_prefetch(emb_table + (size_t)s_id[t + 1] * DD + tid * 3, 0, 3);
    const float r0 = row[tid], r1 = row[tid + 256], r2 = row[tid + 512];
    f0 += r0 * wa; f1 += r1 * wa; f2 += r2 * wa;
#pragma unroll
    for (int s = 0; s < SS; ++s) {
      const float wg = s_g[s][t];
      p0[s] += r0 * wg; p1[s] += r1 * wg; p2[s] += r2 * wg;
    }
  }

  const float denF = fmaxf(teff[b], 1e-9f);
  full[(size_t)b * DD + tid]       = f0 / denF;
  full[(size_t)b * DD + tid + 256] = f1 / denF;
  full[(size_t)b * DD + tid + 512] = f2 / denF;
#pragma unroll
  for (int s = 0; s < SS; ++s) {
    const int sb = s * BB + b;
    const float denP = fmaxf(hsum[sb], 1e-9f);
    pred[(size_t)sb * DD + tid]       = p0[s] / denP;
    pred[(size_t)sb * DD + tid + 256] = p1[s] / denP;
    pred[(size_t)sb * DD + tid + 512] = p2[s] / denP;
  }
}

__global__ __launch_bounds__(256) void k_loss(const float* __restrict__ pred,
                                              const float* __restrict__ full,
                                              float* __restrict__ out) {
  const int tid = threadIdx.x;
  __shared__ float s_red[256];
  float acc = 0.0f;
  const int n = SS * BB * DD;
  for (int i = tid; i < n; i += 256) {
    int rem = i % (BB * DD);
    float d = pred[i] - full[rem];
    acc += d * d;
  }
  s_red[tid] = acc;
  __syncthreads();
  for (int off = 128; off > 0; off >>= 1) {
    if (tid < off) s_red[tid] += s_red[tid + off];
    __syncthreads();
  }
  if (tid == 0) out[0] = s_red[0] / (float)n;
}

// ---------------------------------------------------------------------------
extern "C" void kernel_launch(void* const* d_in, const int* in_sizes, int n_in,
                              void* d_out, int out_size, void* d_ws, size_t ws_size,
                              hipStream_t stream) {
  const int*   ids        = (const int*)d_in[0];
  const float* embeddings = (const float*)d_in[1];
  const float* attn       = (const float*)d_in[2];
  const float* u          = (const float*)d_in[3];
  const float* ln_g       = (const float*)d_in[4];
  const float* ln_b       = (const float*)d_in[5];
  const float* w1         = (const float*)d_in[6];
  const float* b1         = (const float*)d_in[7];
  const float* w2         = (const float*)d_in[8];
  const float* b2         = (const float*)d_in[9];
  const float* emb_table  = (const float*)d_in[10];

  float* z_out = (float*)d_out;                 // [B,T]
  float* g_out = z_out + BB * TT;               // [S,B,T]
  float* recon = g_out + (size_t)SS * BB * TT;  // [1]

  char* ws = (char*)d_ws;
  size_t off = 0;
  auto carve = [&](size_t bytes) -> char* {
    char* p = ws + off;
    off = (off + bytes + 255) & ~(size_t)255;
    return p;
  };
  __bf16* w1bf        = (__bf16*)carve((size_t)HH * DD * 2);
  float*  scores_raw  = (float*)carve((size_t)BB * TT * 4);
  float*  scores_mask = (float*)carve((size_t)BB * TT * 4);
  float*  teff        = (float*)carve(BB * 4);
  float*  hsum        = (float*)carve(SS * BB * 4);
  float*  full        = (float*)carve((size_t)BB * DD * 4);
  float*  pred        = (float*)carve((size_t)SS * BB * DD * 4);

  k_w1cvt<<<(HH * DD + 255) / 256, 256, 0, stream>>>(w1, w1bf, HH * DD);
  k_gemm<<<(BB * TT) / 16, 256, 0, stream>>>(embeddings, attn, ln_g, ln_b,
                                             w1bf, b1, w2, scores_raw);
  k_entmax<<<BB, 256, 0, stream>>>(scores_raw, attn, b2, scores_mask, teff, z_out);
  k_topk<<<SS * BB, 256, 0, stream>>>(scores_mask, attn, u, teff, g_out, hsum);
  k_pool<<<BB, 256, 0, stream>>>(ids, attn, g_out, emb_table, teff, hsum, full, pred);
  k_loss<<<1, 256, 0, stream>>>(pred, full, recon);
}
